// ContentEncoder_10574209483427
// MI455X (gfx1250) — matrix operand order repaired and checked by reading, not compiled, not run on hardware
//
#include <hip/hip_runtime.h>
#include <math.h>

typedef __bf16 bf16;
typedef __attribute__((ext_vector_type(16))) __bf16 v16bf;
typedef __attribute__((ext_vector_type(8)))  __bf16 v8bf;
typedef __attribute__((ext_vector_type(8)))  float  v8f;

#define B_SZ   32
#define T_SZ   1000
#define NTOK   32000      // B*T, multiple of 64
#define C_IN   961
#define K1P    992        // 961 padded to mult of 32
#define C1P    96         // 80 output channels padded to 96 (6 tiles)
#define CH     256
#define K2P    96         // 80 padded to mult of 32
#define CM     768        // CH*3
#define NC     128

static __device__ __forceinline__ int imin(int a, int b) { return a < b ? a : b; }
static __device__ __forceinline__ int imax(int a, int b) { return a > b ? a : b; }

// ---------------------------------------------------------------------------
// WMMA fragment loaders (bf16 16x16x32).
// A (16x32, M=rows): lane m=l&15; half=l>>4 -> K chunks {h*8..+7},{16+h*8..+7}
// B (32x16, N=cols): lane n=l&15 holds column n, K chunks {h*8..+7},{16+h*8..+7}
// C/D (16x16 f32): lane l: col n=l&15, rows m=(l>>4)*8+r
// ---------------------------------------------------------------------------
__device__ __forceinline__ v16bf load_frag_a(const bf16* __restrict__ base,
                                             int row0, int ld, int k0) {
  int lane = threadIdx.x & 31;
  int r = lane & 15, hf = lane >> 4;
  const bf16* p = base + (size_t)(row0 + r) * ld + k0 + hf * 8;
  v8bf c0 = *(const v8bf*)(p);
  v8bf c1 = *(const v8bf*)(p + 16);
  v16bf f;
#pragma unroll
  for (int i = 0; i < 8; ++i) { f[i] = c0[i]; f[i + 8] = c1[i]; }
  return f;
}

__device__ __forceinline__ v16bf load_frag_b(const bf16* __restrict__ base,
                                             int row0, int ld, int k0) {
  int lane = threadIdx.x & 31;
  int r = lane & 15, hf = lane >> 4;
  const bf16* p = base + (size_t)(row0 + r) * ld + k0 + hf * 8;
  v8bf c0 = *(const v8bf*)(p);
  v8bf c1 = *(const v8bf*)(p + 16);
  v16bf f;
#pragma unroll
  for (int i = 0; i < 8; ++i) { f[i] = c0[i]; f[i + 8] = c1[i]; }
  return f;
}

// A-fragment from an LDS-staged (rows x 32) bf16 tile; row0 = m-tile offset
__device__ __forceinline__ v16bf load_frag_a_lds(const bf16* tile, int row0) {
  int lane = threadIdx.x & 31;
  int r = lane & 15, hf = lane >> 4;
  const bf16* p = tile + (row0 + r) * 32 + hf * 8;
  v8bf c0 = *(const v8bf*)(p);
  v8bf c1 = *(const v8bf*)(p + 16);
  v16bf f;
#pragma unroll
  for (int i = 0; i < 8; ++i) { f[i] = c0[i]; f[i + 8] = c1[i]; }
  return f;
}

#define WMMA_BF16(a, b, acc) \
  __builtin_amdgcn_wmma_f32_16x16x32_bf16(false, (a), false, (b), (short)0, (acc), false, false)

__device__ __forceinline__ void store_tile_f32(float* __restrict__ out, int Cout,
                                               int m0, int c0, const v8f& acc,
                                               const float* __restrict__ bias,
                                               const float* __restrict__ resid,
                                               bf16* __restrict__ outB) {
  int lane = threadIdx.x & 31;
  int nn = lane & 15, mh = lane >> 4;
  float bv = bias[c0 + nn];
#pragma unroll
  for (int r = 0; r < 8; ++r) {
    size_t o = (size_t)(m0 + mh * 8 + r) * Cout + c0 + nn;
    float v = acc[r] + bv;
    if (resid) v += resid[o];
    out[o] = v;
    if (outB) outB[o] = (bf16)v;
  }
}

__device__ __forceinline__ void store_tile_bf16(bf16* __restrict__ out, int Cout,
                                                int m0, int c0, const v8f& acc,
                                                const float* __restrict__ bias) {
  int lane = threadIdx.x & 31;
  int nn = lane & 15, mh = lane >> 4;
  float bv = bias[c0 + nn];
#pragma unroll
  for (int r = 0; r < 8; ++r)
    out[(size_t)(m0 + mh * 8 + r) * Cout + c0 + nn] = (bf16)(acc[r] + bv);
}

// ---------------------------------------------------------------------------
// Weight prep kernels
// ---------------------------------------------------------------------------
__global__ void cvt_bf16(const float* __restrict__ in, bf16* __restrict__ out, int n) {
  for (int i = blockIdx.x * blockDim.x + threadIdx.x; i < n; i += gridDim.x * blockDim.x)
    out[i] = (bf16)in[i];
}

// pad rows to rpad-implied count and K to kpad with zeros
__global__ void pad_w_bf16(const float* __restrict__ in, bf16* __restrict__ out,
                           int rin, int kin, int kpad, int total) {
  for (int i = blockIdx.x * blockDim.x + threadIdx.x; i < total; i += gridDim.x * blockDim.x) {
    int r = i / kpad, k = i - r * kpad;
    out[i] = (r < rin && k < kin) ? (bf16)in[(size_t)r * kin + k] : (bf16)0.f;
  }
}

__global__ void pad_b1(const float* __restrict__ in, float* __restrict__ out) {
  int i = threadIdx.x;   // 96 threads
  out[i] = (i < 80) ? in[i] : 0.f;
}

// centroids (CH, NC) f32 -> centT (NC, CH) bf16 + 0.5*||c||^2 per centroid
__global__ void prep_centroids(const float* __restrict__ cent, bf16* __restrict__ centT,
                               float* __restrict__ cn2h) {
  __shared__ float red[CH];
  int n = blockIdx.x;
  int c = threadIdx.x;
  float v = cent[(size_t)c * NC + n];
  centT[(size_t)n * CH + c] = (bf16)v;
  red[c] = v * v;
  __syncthreads();
  for (int s = 128; s > 0; s >>= 1) {
    if (c < s) red[c] += red[c + s];
    __syncthreads();
  }
  if (c == 0) cn2h[n] = 0.5f * red[0];
}

// ---------------------------------------------------------------------------
// input layer 1: Y1(N,96) = spec_tokens(N,961) * w1p^T + b1p
// w1p zero-padded to (96,992): pad outputs fall out of the GEMM for free.
// Block: 256 threads (8 waves), 64-token tile. LDS double-buffered 64x32 tile,
// staged coalesced along t with exactly 8 unguarded passes. Wave w handles
// m-tile (w>>1) and 3 c-tiles at (w&1)*48.
// ---------------------------------------------------------------------------
__global__ void gemm_in1(const float* __restrict__ spec, const bf16* __restrict__ w1p,
                         const float* __restrict__ b1p, bf16* __restrict__ y1) {
  __shared__ __align__(16) bf16 smem[2][64 * 32];
  int wave = threadIdx.x >> 5;
  int n0 = blockIdx.x * 64;

  auto stage = [&](int buf, int k0) {
#pragma unroll
    for (int pass = 0; pass < 8; ++pass) {
      int e = pass * 256 + threadIdx.x;     // 0..2047, no guard
      int r = e & 63, ci = e >> 6;
      int c = k0 + ci;
      int n = n0 + r;
      int b = n / T_SZ, t = n - b * T_SZ;
      int cc = imin(c, C_IN - 1);           // clamp: always in-bounds
      float v = spec[(size_t)b * C_IN * T_SZ + (size_t)cc * T_SZ + t];
      smem[buf][r * 32 + ci] = (bf16)((c < C_IN) ? v : 0.f);
    }
  };

  int wm = (wave >> 1) * 16;                // m-tile row offset in LDS tile
  int cb = (wave & 1) * 48;                 // 3 c-tiles: cb, cb+16, cb+32
  v8f s0 = {}, s1 = {}, s2 = {};
  stage(0, 0);
  for (int step = 0; step < 31; ++step) {
    __syncthreads();
    if (step + 1 < 31) stage((step + 1) & 1, (step + 1) * 32);
    int k0 = step * 32;
    v16bf a  = load_frag_a_lds(smem[step & 1], wm);
    v16bf b0 = load_frag_b(w1p, cb,      K1P, k0);
    v16bf b1 = load_frag_b(w1p, cb + 16, K1P, k0);
    v16bf b2 = load_frag_b(w1p, cb + 32, K1P, k0);
    s0 = WMMA_BF16(a, b0, s0);
    s1 = WMMA_BF16(a, b1, s1);
    s2 = WMMA_BF16(a, b2, s2);
  }
  store_tile_bf16(y1, C1P, n0 + wm, cb,      s0, b1p);
  store_tile_bf16(y1, C1P, n0 + wm, cb + 16, s1, b1p);
  store_tile_bf16(y1, C1P, n0 + wm, cb + 32, s2, b1p);
}

// ---------------------------------------------------------------------------
// generic f32-out GEMM, register blocked 2m x 2c per wave:
//   out(N,Cout=256) = A(N,K)*W(256,K)^T + bias (+resid) [+bf16 mirror]
// grid.x = NTOK/32, block = 256 (8 waves => 256 couts per block)
// ---------------------------------------------------------------------------
__global__ void gemm_f32out(const bf16* __restrict__ A, int K,
                            const bf16* __restrict__ W,
                            const float* __restrict__ bias,
                            const float* __restrict__ resid,
                            float* __restrict__ out,
                            bf16* __restrict__ outB) {
  int wave = threadIdx.x >> 5;
  int c0 = wave * 32;
  int m0 = blockIdx.x * 32;
  v8f a00 = {}, a01 = {}, a10 = {}, a11 = {};
  for (int k0 = 0; k0 < K; k0 += 32) {
    v16bf af0 = load_frag_a(A, m0,      K, k0);
    v16bf af1 = load_frag_a(A, m0 + 16, K, k0);
    v16bf bf0 = load_frag_b(W, c0,      K, k0);
    v16bf bf1 = load_frag_b(W, c0 + 16, K, k0);
    a00 = WMMA_BF16(af0, bf0, a00);
    a01 = WMMA_BF16(af0, bf1, a01);
    a10 = WMMA_BF16(af1, bf0, a10);
    a11 = WMMA_BF16(af1, bf1, a11);
  }
  store_tile_f32(out, CH, m0,      c0,      a00, bias, resid, outB);
  store_tile_f32(out, CH, m0,      c0 + 16, a01, bias, resid, outB);
  store_tile_f32(out, CH, m0 + 16, c0,      a10, bias, resid, outB);
  store_tile_f32(out, CH, m0 + 16, c0 + 16, a11, bias, resid, outB);
}

// output layer: z(N,256) bf16 = xb(N,256)*wout^T + b_out   (2m x 2c per wave)
__global__ void gemm_outlayer(const bf16* __restrict__ A, const bf16* __restrict__ W,
                              const float* __restrict__ bias, bf16* __restrict__ out) {
  int wave = threadIdx.x >> 5;
  int c0 = wave * 32;
  int m0 = blockIdx.x * 32;
  v8f a00 = {}, a01 = {}, a10 = {}, a11 = {};
  for (int k0 = 0; k0 < CH; k0 += 32) {
    v16bf af0 = load_frag_a(A, m0,      CH, k0);
    v16bf af1 = load_frag_a(A, m0 + 16, CH, k0);
    v16bf bf0 = load_frag_b(W, c0,      CH, k0);
    v16bf bf1 = load_frag_b(W, c0 + 16, CH, k0);
    a00 = WMMA_BF16(af0, bf0, a00);
    a01 = WMMA_BF16(af0, bf1, a01);
    a10 = WMMA_BF16(af1, bf0, a10);
    a11 = WMMA_BF16(af1, bf1, a11);
  }
  store_tile_bf16(out, CH, m0,      c0,      a00, bias);
  store_tile_bf16(out, CH, m0,      c0 + 16, a01, bias);
  store_tile_bf16(out, CH, m0 + 16, c0,      a10, bias);
  store_tile_bf16(out, CH, m0 + 16, c0 + 16, a11, bias);
}

// ---------------------------------------------------------------------------
// fused: scores(32,128) in LDS -> parallel top-4 -> centroid mean -> out
// scores = 0.5*||c||^2 - z.c  (rank-equivalent to squared distance)
// block 256 (8 waves), 32-token tile; wave w: 16 centroids, 2 m-tiles.
// After barrier wave w selects top-4 for tokens w*4..w*4+3 (lane-parallel).
// ---------------------------------------------------------------------------
__global__ void gemm_scores_topk(const bf16* __restrict__ z, const bf16* __restrict__ cT,
                                 const float* __restrict__ cn2h,
                                 const float* __restrict__ cent,
                                 float* __restrict__ out) {
  __shared__ float ssc[32][NC + 4];
  int wave = threadIdx.x >> 5, lane = threadIdx.x & 31;
  int c0 = wave * 16;
  int m0 = blockIdx.x * 32;
  v8f a0 = {}, a1 = {};
  for (int k0 = 0; k0 < CH; k0 += 32) {
    v16bf af0 = load_frag_a(z, m0,      CH, k0);
    v16bf af1 = load_frag_a(z, m0 + 16, CH, k0);
    v16bf bf0 = load_frag_b(cT, c0, CH, k0);
    a0 = WMMA_BF16(af0, bf0, a0);
    a1 = WMMA_BF16(af1, bf0, a1);
  }
  int nn = lane & 15, mh = lane >> 4;
  float cv = cn2h[c0 + nn];
#pragma unroll
  for (int r = 0; r < 8; ++r) {
    ssc[mh * 8 + r][c0 + nn]      = cv - a0[r];
    ssc[16 + mh * 8 + r][c0 + nn] = cv - a1[r];
  }
  __syncthreads();

#pragma unroll
  for (int tk = 0; tk < 4; ++tk) {
    int m = wave * 4 + tk;
    float v[4];
#pragma unroll
    for (int q = 0; q < 4; ++q) v[q] = ssc[m][lane * 4 + q];
    int n = m0 + m;
    int b = n / T_SZ, t = n - b * T_SZ;
    int chosen[4];
#pragma unroll
    for (int j = 0; j < 4; ++j) {
      float bl = v[0];
      int bi = lane * 4;
#pragma unroll
      for (int q = 1; q < 4; ++q)
        if (v[q] < bl) { bl = v[q]; bi = lane * 4 + q; }
#pragma unroll
      for (int off = 16; off > 0; off >>= 1) {
        float ov = __shfl_xor(bl, off, 32);
        int   oi = __shfl_xor(bi, off, 32);
        if (ov < bl || (ov == bl && oi < bi)) { bl = ov; bi = oi; }
      }
      chosen[j] = bi;
      if ((bi >> 2) == lane) v[bi & 3] = 3.4e38f;
    }
#pragma unroll
    for (int q = 0; q < 8; ++q) {
      int ch = lane + 32 * q;
      const float* cr = cent + (size_t)ch * NC;
      float s = 0.25f * (cr[chosen[0]] + cr[chosen[1]] + cr[chosen[2]] + cr[chosen[3]]);
      out[(size_t)(b * CH + ch) * T_SZ + t] = s;
    }
  }
}

// ---------------------------------------------------------------------------
// ChannelNorm: one wave32 per token (unbiased std, sigma += 1e-4)
// ---------------------------------------------------------------------------
__global__ void channel_norm(const float* __restrict__ x, const float* __restrict__ scale,
                             const float* __restrict__ shift, float* __restrict__ h) {
  int wave = threadIdx.x >> 5, lane = threadIdx.x & 31;
  int n = blockIdx.x * 8 + wave;
  const float* xp = x + (size_t)n * CH;
  float v[8], s = 0.f, ss = 0.f;
#pragma unroll
  for (int q = 0; q < 8; ++q) {
    v[q] = xp[lane + 32 * q];
    s += v[q];
    ss += v[q] * v[q];
  }
#pragma unroll
  for (int off = 16; off > 0; off >>= 1) {
    s += __shfl_xor(s, off, 32);
    ss += __shfl_xor(ss, off, 32);
  }
  float mu = s * (1.f / 256.f);
  float var = (ss - 256.f * mu * mu) * (1.f / 255.f);
  float inv = 1.f / (sqrtf(fmaxf(var, 0.f)) + 1e-4f);
#pragma unroll
  for (int q = 0; q < 8; ++q) {
    int c = lane + 32 * q;
    h[(size_t)n * CH + c] = (v[q] - mu) * inv * scale[c] + shift[c];
  }
}

// ---------------------------------------------------------------------------
// Depthwise dilated conv (k=7, groups=256) + exact GELU -> bf16
// one thread per (token, group): 7 shared taps -> 3 outputs
// ---------------------------------------------------------------------------
__global__ void dwconv_gelu(const float* __restrict__ h, const float* __restrict__ w,
                            const float* __restrict__ bias, bf16* __restrict__ h2, int dil) {
  int idx = blockIdx.x * blockDim.x + threadIdx.x;   // NTOK*CH threads
  int n = idx >> 8, ch = idx & 255;
  int b = n / T_SZ, t = n - b * T_SZ;
  float hv[7];
#pragma unroll
  for (int k = 0; k < 7; ++k) {
    int tt = t + (k - 3) * dil;
    bool ok = (unsigned)tt < (unsigned)T_SZ;
    int ttc = imin(imax(tt, 0), T_SZ - 1);
    float vv = h[(size_t)(b * T_SZ + ttc) * CH + ch];
    hv[k] = ok ? vv : 0.f;
  }
#pragma unroll
  for (int j = 0; j < 3; ++j) {
    int co = ch * 3 + j;
    float s = bias[co];
#pragma unroll
    for (int k = 0; k < 7; ++k) s += hv[k] * w[co * 7 + k];
    float g = 0.5f * s * (1.f + erff(s * 0.70710678118654752f));
    h2[(size_t)n * CM + co] = (bf16)g;
  }
}

// ---------------------------------------------------------------------------
extern "C" void kernel_launch(void* const* d_in, const int* in_sizes, int n_in,
                              void* d_out, int out_size, void* d_ws, size_t ws_size,
                              hipStream_t stream) {
  const float* spec   = (const float*)d_in[0];
  const float* w_in1  = (const float*)d_in[1];
  const float* b_in1  = (const float*)d_in[2];
  const float* w_in2  = (const float*)d_in[3];
  const float* b_in2  = (const float*)d_in[4];
  const float* nscale = (const float*)d_in[5];
  const float* nshift = (const float*)d_in[6];
  const float* dw_w   = (const float*)d_in[7];
  const float* dw_b   = (const float*)d_in[8];
  const float* pw_w   = (const float*)d_in[9];
  const float* pw_b   = (const float*)d_in[10];
  const float* w_out  = (const float*)d_in[11];
  const float* b_out  = (const float*)d_in[12];
  const float* cent   = (const float*)d_in[13];
  float* out = (float*)d_out;

  char* ws = (char*)d_ws;
  size_t off = 0;
  auto take = [&](size_t bytes) -> char* {
    char* p = ws + off;
    off += (bytes + 255) & ~(size_t)255;
    return p;
  };
  bf16*  w1p   = (bf16*)take((size_t)C1P * K1P * 2);      // (96,992) zero-padded
  float* b1p   = (float*)take((size_t)C1P * 4);
  bf16*  w2p   = (bf16*)take((size_t)CH * K2P * 2);
  bf16*  pwb   = (bf16*)take((size_t)4 * CH * CM * 2);
  bf16*  woutb = (bf16*)take((size_t)CH * CH * 2);
  bf16*  centT = (bf16*)take((size_t)NC * CH * 2);
  float* cn2h  = (float*)take((size_t)NC * 4);
  float* x     = (float*)take((size_t)NTOK * CH * 4);
  float* h     = (float*)take((size_t)NTOK * CH * 4);
  char*  big   = take((size_t)NTOK * CM * 2);             // h2 / y1 reuse
  bf16*  xb    = (bf16*)take((size_t)NTOK * CH * 2);      // bf16 mirror of final x
  bf16*  h2 = (bf16*)big;
  bf16*  y1 = (bf16*)big;
  bf16*  z  = (bf16*)x;                                   // x dead once xb exists

  // --- weight prep ---
  pad_w_bf16<<<(C1P * K1P + 255) / 256, 256, 0, stream>>>(w_in1, w1p, 80, C_IN, K1P, C1P * K1P);
  pad_b1<<<1, C1P, 0, stream>>>(b_in1, b1p);
  pad_w_bf16<<<(CH * K2P + 255) / 256, 256, 0, stream>>>(w_in2, w2p, CH, 80, K2P, CH * K2P);
  cvt_bf16<<<1024, 256, 0, stream>>>(pw_w, pwb, 4 * CH * CM);
  cvt_bf16<<<256, 256, 0, stream>>>(w_out, woutb, CH * CH);
  prep_centroids<<<NC, CH, 0, stream>>>(cent, centT, cn2h);

  // --- input layers ---
  gemm_in1<<<NTOK / 64, 256, 0, stream>>>(spec, w1p, b1p, y1);
  gemm_f32out<<<NTOK / 32, 256, 0, stream>>>(y1, K2P, w2p, b_in2, nullptr, x, nullptr);

  // --- 4 ResBlocks ---
  int dil = 1;
  for (int l = 0; l < 4; ++l) {
    channel_norm<<<NTOK / 8, 256, 0, stream>>>(x, nscale + l * CH, nshift + l * CH, h);
    dwconv_gelu<<<(NTOK * CH) / 256, 256, 0, stream>>>(h, dw_w + (size_t)l * CM * 7,
                                                       dw_b + l * CM, h2, dil);
    gemm_f32out<<<NTOK / 32, 256, 0, stream>>>(h2, CM, pwb + (size_t)l * CH * CM,
                                               pw_b + l * CH, x, x,
                                               (l == 3) ? xb : nullptr);
    dil *= 3;
  }

  // --- output layer + fused VQ ---
  gemm_outlayer<<<NTOK / 32, 256, 0, stream>>>(xb, woutb, b_out, z);
  gemm_scores_topk<<<NTOK / 32, 256, 0, stream>>>(z, centT, cn2h, cent, out);
}
